// AttentionHead_79843442032739
// MI455X (gfx1250) — compile-verified, matrix operand order checked by reading, not run on hardware
//
#include <hip/hip_runtime.h>
#include <hip/hip_bf16.h>

// ---------------------------------------------------------------------------
// CDNA5 (gfx1250) wave32 WMMA attention pipeline, bf16 inputs / f32 accum.
// Scores kernel stages K tiles through LDS with the Tensor Data Mover (TDM).
// ---------------------------------------------------------------------------

typedef __attribute__((ext_vector_type(16))) __bf16 v16bf;
typedef __attribute__((ext_vector_type(8)))  __bf16 v8bf;
typedef __attribute__((ext_vector_type(8)))  float  v8f;
typedef __attribute__((ext_vector_type(4)))  unsigned int u32x4;
typedef __attribute__((ext_vector_type(8)))  int i32x8;
typedef __attribute__((ext_vector_type(4)))  int i32x4;

#define B_  4
#define S_  2048
#define D_  1024
#define NEG_SENT (-1.0e30f)

// WMMA 16x16x32 bf16 -> f32 accumulate.
__device__ __forceinline__ v8f wmma_bf16(v16bf a, v16bf b, v8f c) {
    return __builtin_amdgcn_wmma_f32_16x16x32_bf16(false, a, false, b,
                                                   (short)0, c, false, false);
}

// Load one 16x32 (16-bit) fragment from a row-major tile with K contiguous.
// ISA 16-bit A 16x32 layout: lane l holds row (l&15); lanes 0-15 hold
// K={0..7,16..23}, lanes 16-31 hold K={8..15,24..31} -> two 16B chunks/lane.
__device__ __forceinline__ v16bf load_frag(const __bf16* __restrict__ base,
                                           int row_stride) {
    int lane = threadIdx.x & 31;
    int r = lane & 15;
    int h = lane >> 4;
    const __bf16* p = base + (size_t)r * row_stride + h * 8;
    v8bf lo = *(const v8bf*)(p);
    v8bf hi = *(const v8bf*)(p + 16);
    return __builtin_shufflevector(lo, hi, 0, 1, 2, 3, 4, 5, 6, 7,
                                   8, 9, 10, 11, 12, 13, 14, 15);
}

// Same fragment gather, but from an LDS-resident tile (ds_load_b128 path).
__device__ __forceinline__ v16bf load_frag_lds(const __bf16* base,
                                               int row_stride) {
    int lane = threadIdx.x & 31;
    int r = lane & 15;
    int h = lane >> 4;
    const __bf16* p = base + r * row_stride + h * 8;
    v8bf lo = *(const v8bf*)(p);
    v8bf hi = *(const v8bf*)(p + 16);
    return __builtin_shufflevector(lo, hi, 0, 1, 2, 3, 4, 5, 6, 7,
                                   8, 9, 10, 11, 12, 13, 14, 15);
}

// ---------------------------------------------------------------------------
// TDM: issue a 2D tensor_load_to_lds of a 64-row x 128-col bf16 tile from a
// row-major [S_ x D_] tensor (row stride D_ elements) into LDS at lds_off.
// D# layout per CDNA5 ISA ch.8 (count=1, type=2 "image", data_size=2 bytes).
// ---------------------------------------------------------------------------
__device__ __forceinline__ void tdm_load_tile_64x128(unsigned long long gaddr_bytes,
                                                     unsigned lds_off) {
    unsigned ga_lo = (unsigned)__builtin_amdgcn_readfirstlane((int)(unsigned)gaddr_bytes);
    unsigned ga_hi = (unsigned)__builtin_amdgcn_readfirstlane((int)(unsigned)(gaddr_bytes >> 32));
    unsigned loff  = (unsigned)__builtin_amdgcn_readfirstlane((int)lds_off);

    u32x4 g0;
    g0.x = 1u;                                      // count=1, user mode
    g0.y = loff;                                    // lds_addr (bytes)
    g0.z = ga_lo;                                   // global_addr[31:0]
    g0.w = (ga_hi & 0x01FFFFFFu) | (2u << 30);      // global_addr[56:32] | type=2

    i32x8 g1;
    g1.s0 = 0x00010000;                             // mask=0, data_size=1 (2B)
    g1.s1 = (int)((D_ & 0xFFFF) << 16);             // tensor_dim0[15:0]
    g1.s2 = (int)((D_ >> 16) | ((S_ & 0xFFFF) << 16));   // dim0 hi | dim1 lo
    g1.s3 = (int)((S_ >> 16) | (128 << 16));        // dim1 hi | tile_dim0=128
    g1.s4 = 64;                                     // tile_dim1=64, tile_dim2=0
    g1.s5 = D_;                                     // tensor_dim0_stride lo32
    g1.s6 = 0;                                      // stride hi | stride1 lo
    g1.s7 = 0;

    i32x4 z4 = {0, 0, 0, 0};
#if __clang_major__ >= 23
    i32x8 z8 = {0, 0, 0, 0, 0, 0, 0, 0};
    __builtin_amdgcn_tensor_load_to_lds(g0, g1, z4, z4, z8, 0);
#else
    __builtin_amdgcn_tensor_load_to_lds(g0, g1, z4, z4, 0);
#endif
}

// ---------------------------------------------------------------------------
// 1) elementwise fp32 -> bf16
// ---------------------------------------------------------------------------
__global__ void cvt_f32_bf16(const float* __restrict__ in,
                             __bf16* __restrict__ out, size_t n) {
    size_t i = (size_t)blockIdx.x * blockDim.x + threadIdx.x;
    if (i < n) out[i] = (__bf16)in[i];
}

// W [D x D] fp32 row-major -> Wt [D x D] bf16 with Wt[n][k] = W[k][n]
__global__ void transpose_w(const float* __restrict__ W,
                            __bf16* __restrict__ Wt) {
    int idx = blockIdx.x * blockDim.x + threadIdx.x;
    if (idx < D_ * D_) {
        int k = idx >> 10;
        int n = idx & (D_ - 1);
        Wt[(size_t)n * D_ + k] = (__bf16)W[idx];
    }
}

// ---------------------------------------------------------------------------
// 2) Projection GEMM: Out[M=8192,N=1024] = X[M,K] * W (via Wt[N][K]).
//    Block = 8 waves (4x2), each wave computes a 64x64 tile via 4x4 WMMA
//    accumulators -> 1 fragment load per WMMA (vs 2 at 32x32).
// ---------------------------------------------------------------------------
template <int TRANSPOSE_STORE>
__global__ void __launch_bounds__(256)
proj_gemm(const __bf16* __restrict__ X,
          const __bf16* __restrict__ Wt,
          __bf16* __restrict__ Out) {
    const int N = D_, K = D_;
    int wave = threadIdx.x >> 5;
    int lane = threadIdx.x & 31;
    int wm = wave & 3, wn = wave >> 2;              // 4x2 wave grid
    int m0 = blockIdx.x * 256 + wm * 64;
    int n0 = blockIdx.y * 128 + wn * 64;

    v8f acc[4][4] = {};
#pragma unroll 1
    for (int k0 = 0; k0 < K; k0 += 32) {
        __builtin_prefetch(Wt + (size_t)n0 * K + k0 + 64, 0, 1);
        v16bf a[4];
#pragma unroll
        for (int i = 0; i < 4; ++i)
            a[i] = load_frag(X + (size_t)(m0 + 16 * i) * K + k0, K);
#pragma unroll
        for (int j = 0; j < 4; ++j) {
            v16bf bf = load_frag(Wt + (size_t)(n0 + 16 * j) * K + k0, K);
#pragma unroll
            for (int i = 0; i < 4; ++i)
                acc[i][j] = wmma_bf16(a[i], bf, acc[i][j]);
        }
    }

    int r = lane & 15, h = lane >> 4;
#pragma unroll
    for (int i = 0; i < 4; ++i)
#pragma unroll
        for (int j = 0; j < 4; ++j)
#pragma unroll
            for (int rr = 0; rr < 8; ++rr) {
                int m = m0 + i * 16 + rr + 8 * h;   // C layout: M = r + 8*half
                int n = n0 + j * 16 + r;            //           N = lane&15
                __bf16 v = (__bf16)acc[i][j][rr];
                if (TRANSPOSE_STORE) {
                    int b = m >> 11;
                    int s = m & (S_ - 1);
                    Out[((size_t)b * N + n) * S_ + s] = v;
                } else {
                    Out[(size_t)m * N + n] = v;
                }
            }
}

// ---------------------------------------------------------------------------
// 3) Scores: raw = (Q K^T)/sqrt(D), causal mask (finite sentinel), online
//    per-lane (max,sum). One wave owns 16 query rows; the 4 waves of a block
//    own 4 consecutive q-tiles and SHARE TDM-staged K tiles in LDS:
//    wave0 DMAs 64-key x 128-feature chunks (double buffered, 32KB LDS),
//    everyone consumes B-fragments via ds_load_b128.
// ---------------------------------------------------------------------------
__global__ void __launch_bounds__(128)
scores_kernel(const __bf16* __restrict__ Qb,
              const __bf16* __restrict__ Kb,
              float* __restrict__ P32,
              float* __restrict__ rowM,
              float* __restrict__ rowL) {
    __shared__ __bf16 shK[2][64 * 128];             // 2 x 16KB double buffer

    int wave = threadIdx.x >> 5;
    int lane = threadIdx.x & 31;
    int r = lane & 15, h = lane >> 4;
    int gwave = blockIdx.x * 4 + wave;
    int b = gwave >> 7;                             // 128 q-tiles per batch
    int qt = (gwave & 127) << 4;

    const __bf16* Q = Qb + ((size_t)b * S_ + qt) * D_;
    float* Prow = P32 + ((size_t)b * S_ + qt) * S_;
    const float scale = 0.03125f;                   // 1/sqrt(1024)

    // Block-shared causal k-range (from the largest q-tile in this block).
    int qt_blk = ((blockIdx.x * 4 + 3) & 127) << 4;
    int kmax_blk = ((qt_blk + 16 + 63) >> 6) << 6;
    int nchunks = (kmax_blk >> 6) << 3;             // 8 e-chunks per 64 keys

    unsigned long long kbase =
        (unsigned long long)(const void*)(Kb + (size_t)b * S_ * D_);

    float m_run[8], l_run[8];
#pragma unroll
    for (int i = 0; i < 8; ++i) { m_run[i] = NEG_SENT; l_run[i] = 0.0f; }

    v8f acc[4] = {};

    if (wave == 0) {
        tdm_load_tile_64x128(kbase, (unsigned)(size_t)&shK[0][0]);
    }

#pragma unroll 1
    for (int c = 0; c < nchunks; ++c) {
        if (wave == 0) __builtin_amdgcn_s_wait_tensorcnt(0);
        __syncthreads();                            // staged chunk visible
        if (wave == 0 && (c + 1) < nchunks) {
            int cn = c + 1;
            int k0n = (cn >> 3) << 6;
            int ecn = (cn & 7) << 7;
            unsigned long long ga =
                kbase + ((unsigned long long)((size_t)k0n * D_ + ecn) << 1);
            tdm_load_tile_64x128(ga, (unsigned)(size_t)&shK[cn & 1][0]);
        }

        int k0 = (c >> 3) << 6;
        int ec = (c & 7) << 7;
        const __bf16* lb = &shK[c & 1][0];
#pragma unroll
        for (int t = 0; t < 4; ++t) {
            int e = ec + t * 32;
            v16bf a = load_frag(Q + e, D_);
#pragma unroll
            for (int j = 0; j < 4; ++j) {
                v16bf bf = load_frag_lds(lb + (16 * j) * 128 + t * 32, 128);
                acc[j] = wmma_bf16(a, bf, acc[j]);
            }
        }

        if ((c & 7) == 7) {                         // finished 64-key chunk
#pragma unroll
            for (int j = 0; j < 4; ++j) {
#pragma unroll
                for (int rr = 0; rr < 8; ++rr) {
                    int q = qt + rr + 8 * h;
                    int k = k0 + 16 * j + r;
                    float s = acc[j][rr] * scale;
                    if (k > q) s = NEG_SENT;        // causal (finite)
                    float mo = m_run[rr];
                    float mn = fmaxf(mo, s);
                    l_run[rr] = l_run[rr] * __expf(mo - mn) + __expf(s - mn);
                    m_run[rr] = mn;
                    Prow[(size_t)(rr + 8 * h) * S_ + k] = s;
                }
                acc[j] = (v8f){};
            }
        }
        __syncthreads();                            // safe to overwrite buffer
    }

    // merge (m,l) across the 16 lanes of each half-wave
#pragma unroll
    for (int rr = 0; rr < 8; ++rr) {
        float m = m_run[rr], l = l_run[rr];
        for (int off = 1; off < 16; off <<= 1) {
            float m2 = __shfl_xor(m, off, 16);
            float l2 = __shfl_xor(l, off, 16);
            float mn = fmaxf(m, m2);
            l = l * __expf(m - mn) + l2 * __expf(m2 - mn);
            m = mn;
        }
        if (r == 0) {
            int q = qt + rr + 8 * h;
            rowM[(size_t)b * S_ + q] = m;
            rowL[(size_t)b * S_ + q] = l;
        }
    }
}

// ---------------------------------------------------------------------------
// 4) Normalize raw scores -> bf16 probabilities (written region only).
// ---------------------------------------------------------------------------
__global__ void normalize_kernel(const float* __restrict__ P32,
                                 const float* __restrict__ rowM,
                                 const float* __restrict__ rowL,
                                 __bf16* __restrict__ Pb) {
    int row = blockIdx.x;                           // B_*S_ rows
    int q = row & (S_ - 1);
    int qt = q & ~15;
    int kmax = ((qt + 16 + 63) >> 6) << 6;
    float M = rowM[row];
    float Li = 1.0f / rowL[row];
    const float* src = P32 + (size_t)row * S_;
    __bf16* dst = Pb + (size_t)row * S_;
    for (int k = threadIdx.x; k < kmax; k += blockDim.x)
        dst[k] = (__bf16)(__expf(src[k] - M) * Li);
}

// ---------------------------------------------------------------------------
// 5) Output GEMM: out[b][q][e] = sum_k P[b][q][k] * V[b][k][e] via Vt[b][e][k].
//    Wave computes 16 q-rows x 64 e-cols (1x4 tiles -> each P fragment is
//    reused by 4 WMMAs); k-loop honors causality.
// ---------------------------------------------------------------------------
__global__ void __launch_bounds__(256)
out_gemm(const __bf16* __restrict__ Pb,
         const __bf16* __restrict__ Vt,
         float* __restrict__ Out) {
    int gwave = (blockIdx.x * blockDim.x + threadIdx.x) >> 5;
    int b = gwave >> 11;                            // (S_/16)*(D_/64)=2048/batch
    int rem = gwave & 2047;
    int qt = (rem >> 4) << 4;
    int e0 = (rem & 15) << 6;
    int lane = threadIdx.x & 31;

    const __bf16* P = Pb + ((size_t)b * S_ + qt) * S_;
    const __bf16* V0 = Vt + ((size_t)b * D_ + e0) * S_;

    v8f acc[4] = {};
    int kend = ((qt + 16 + 31) >> 5) << 5;
#pragma unroll 1
    for (int k0 = 0; k0 < kend; k0 += 32) {
        v16bf a = load_frag(P + k0, S_);
#pragma unroll
        for (int j = 0; j < 4; ++j) {
            v16bf bf = load_frag(V0 + (size_t)(16 * j) * S_ + k0, S_);
            acc[j] = wmma_bf16(a, bf, acc[j]);
        }
    }

    int r = lane & 15, h = lane >> 4;
#pragma unroll
    for (int j = 0; j < 4; ++j)
#pragma unroll
        for (int rr = 0; rr < 8; ++rr) {
            int q = qt + rr + 8 * h;
            Out[((size_t)b * S_ + q) * D_ + e0 + 16 * j + r] = acc[j][rr];
        }
}

// ---------------------------------------------------------------------------
extern "C" void kernel_launch(void* const* d_in, const int* in_sizes, int n_in,
                              void* d_out, int out_size, void* d_ws, size_t ws_size,
                              hipStream_t stream) {
    const float* Xk = (const float*)d_in[0];   // inputs_for_keys
    const float* Xv = (const float*)d_in[1];   // inputs_for_values
    const float* Xq = (const float*)d_in[2];   // inputs_for_queries
    const float* WK = (const float*)d_in[3];
    const float* WQ = (const float*)d_in[4];
    const float* WV = (const float*)d_in[5];
    float* Out = (float*)d_out;

    char* ws = (char*)d_ws;
    const size_t MB = 1024ull * 1024ull;
    __bf16* Xqb = (__bf16*)(ws + 0 * MB);
    __bf16* Xkb = (__bf16*)(ws + 16 * MB);
    __bf16* Xvb = (__bf16*)(ws + 32 * MB);
    __bf16* Wtq = (__bf16*)(ws + 48 * MB);
    __bf16* Wtk = (__bf16*)(ws + 50 * MB);
    __bf16* Wtv = (__bf16*)(ws + 52 * MB);
    __bf16* Qb  = (__bf16*)(ws + 54 * MB);
    __bf16* Kb  = (__bf16*)(ws + 70 * MB);
    __bf16* Vt  = (__bf16*)(ws + 86 * MB);     // V transposed per batch [D][S]
    float*  P32 = (float*)(ws + 102 * MB);     // raw scores
    float*  rM  = (float*)(ws + 166 * MB);
    float*  rL  = (float*)(ws + 166 * MB + 64 * 1024);
    __bf16* Pb  = (__bf16*)(ws + 167 * MB);    // probabilities

    const size_t nX = (size_t)B_ * S_ * D_;

    cvt_f32_bf16<<<(unsigned)((nX + 255) / 256), 256, 0, stream>>>(Xq, Xqb, nX);
    cvt_f32_bf16<<<(unsigned)((nX + 255) / 256), 256, 0, stream>>>(Xk, Xkb, nX);
    cvt_f32_bf16<<<(unsigned)((nX + 255) / 256), 256, 0, stream>>>(Xv, Xvb, nX);

    transpose_w<<<(D_ * D_ + 255) / 256, 256, 0, stream>>>(WQ, Wtq);
    transpose_w<<<(D_ * D_ + 255) / 256, 256, 0, stream>>>(WK, Wtk);
    transpose_w<<<(D_ * D_ + 255) / 256, 256, 0, stream>>>(WV, Wtv);

    dim3 pgrid(B_ * S_ / 256, D_ / 128);
    proj_gemm<0><<<pgrid, 256, 0, stream>>>(Xqb, Wtq, Qb);
    proj_gemm<0><<<pgrid, 256, 0, stream>>>(Xkb, Wtk, Kb);
    proj_gemm<1><<<pgrid, 256, 0, stream>>>(Xvb, Wtv, Vt);

    scores_kernel<<<(B_ * (S_ / 16)) / 4, 128, 0, stream>>>(Qb, Kb, P32, rM, rL);

    normalize_kernel<<<B_ * S_, 256, 0, stream>>>(P32, rM, rL, Pb);

    // out = P @ V : 8192 waves, 8 waves/block
    out_gemm<<<(B_ * (S_ / 16) * (D_ / 64)) / 8, 256, 0, stream>>>(Pb, Vt, Out);
}
